// MPNNGraphRegression_36885179138612
// MI455X (gfx1250) — matrix operand order repaired and checked by reading, not compiled, not run on hardware
//
#include <hip/hip_runtime.h>
#include <hip/hip_bf16.h>

typedef __attribute__((ext_vector_type(16))) _Float16 v16h;
typedef __attribute__((ext_vector_type(8)))  float    v8f;

#define EWAVES 4
#define NWAVES 4

// ---------------------------------------------------------------------------
// WMMA helpers (CDNA5, wave32). Layouts per CDNA5 ISA §7.12.2:
//   A (16x32 f16): lane<16 -> M=lane, K = {0..7, 16..23}; lane>=16 -> +8 on K
//   B (32x16 f16): N = lane&15, K = (half_idx&7) + (lane>=16 ? 8 : 0) + (half_idx>=8 ? 16 : 0)
//   C/D (16x16 f32): vgpr r -> M = r + (lane>=16 ? 8 : 0), N = lane&15
// ---------------------------------------------------------------------------
__device__ __forceinline__ v8f wmma16(v16h a, v16h b, v8f c) {
  return __builtin_amdgcn_wmma_f32_16x16x32_f16(
      /*neg_a=*/false, a, /*neg_b=*/false, b,
      /*c_mod=*/(short)0, c, /*reuse_a=*/false, /*reuse_b=*/false);
}

// A fragment (row m = lane&15) from a 32-col slice of an f32 row (agg path).
__device__ __forceinline__ v16h afrag_f32(const float* __restrict__ row,
                                          int kbase, int lane) {
  const int off = (lane & 16) ? 8 : 0;
  const float* p0 = row + kbase + off;
  const float* p1 = row + kbase + 16 + off;
  v16h a;
#pragma unroll
  for (int i = 0; i < 8; ++i) {
    a[i]     = (_Float16)p0[i];
    a[i + 8] = (_Float16)p1[i];
  }
  return a;
}

// A fragment from an f16 row (global h buffers): two 16B contiguous chunks.
__device__ __forceinline__ v16h afrag_f16g(const _Float16* __restrict__ row,
                                           int kbase, int lane) {
  const int off = (lane & 16) ? 8 : 0;
  v16h a;
#pragma unroll
  for (int i = 0; i < 8; ++i) {
    a[i]     = row[kbase + off + i];
    a[i + 8] = row[kbase + 16 + off + i];
  }
  return a;
}

// A fragment from an f16 row staged in LDS.
__device__ __forceinline__ v16h afrag_lds(const _Float16* __restrict__ row,
                                          int kbase, int lane) {
  const int off = (lane & 16) ? 8 : 0;
  v16h a;
#pragma unroll
  for (int i = 0; i < 8; ++i) {
    a[i]     = row[kbase + off + i];
    a[i + 8] = row[kbase + 16 + off + i];
  }
  return a;
}

// Pre-packed B fragments: [kt][nt][lane][16] halves, 32B per lane.
__device__ __forceinline__ v16h bfrag(const _Float16* __restrict__ packed,
                                      int kt, int nt, int lane) {
  return *(const v16h*)(packed + (((kt * 4 + nt) * 32) + lane) * 16);
}

__device__ __forceinline__ void acc_zero(v8f* acc) {
#pragma unroll
  for (int t = 0; t < 4; ++t)
#pragma unroll
    for (int i = 0; i < 8; ++i) acc[t][i] = 0.f;
}

// ---------------------------------------------------------------------------
// Weight packer: f32 [L, Klog, 64] -> f16 fragment layout
// [l][ktile][ntile][lane][16], zero-padding K up to Ktiles*32.
// ---------------------------------------------------------------------------
__global__ void pack_weights(const float* __restrict__ W, _Float16* __restrict__ out,
                             int L, int Klog, int Ktiles) {
  const int total = L * Ktiles * 2048;  // 4 ntiles * 32 lanes * 16 halves
  int idx = blockIdx.x * blockDim.x + threadIdx.x;
  if (idx >= total) return;
  const int hh   = idx & 15;
  const int lane = (idx >> 4) & 31;
  const int nt   = (idx >> 9) & 3;
  const int rest = idx >> 11;
  const int kt   = rest % Ktiles;
  const int l    = rest / Ktiles;
  const int k = kt * 32 + (hh & 7) + ((lane & 16) ? 8 : 0) + ((hh & 8) ? 16 : 0);
  const int n = nt * 16 + (lane & 15);
  out[idx] = (k < Klog) ? (_Float16)W[((long long)l * Klog + k) * 64 + n]
                        : (_Float16)0.f;
}

__global__ void zero_f32(float* __restrict__ p, long long n) {
  long long i = (long long)blockIdx.x * blockDim.x + threadIdx.x;
  if (i < n) p[i] = 0.f;
}

__global__ void f32_to_f16(const float* __restrict__ src, _Float16* __restrict__ dst,
                           long long n) {
  long long i = (long long)blockIdx.x * blockDim.x + threadIdx.x;
  if (i < n) dst[i] = (_Float16)src[i];
}

// ---------------------------------------------------------------------------
// Edge message kernel: per wave, a tile of 16 edges. h is stored f16.
//   m_in = [h[dst](64) | h[src](64) | edge_attr(16) | pad(16)]  (K padded 160)
//   m = relu(m_in @ W1 + b1) @ W2 + b2 ; f32 atomic scatter-add into agg[dst].
// ---------------------------------------------------------------------------
__global__ void __launch_bounds__(EWAVES * 32)
edge_msg_kernel(const _Float16* __restrict__ h,
                const long long* __restrict__ eidx,   // [2, E]
                const float* __restrict__ eattr,      // [E, 16]
                const _Float16* __restrict__ w1f, const float* __restrict__ b1,
                const _Float16* __restrict__ w2f, const float* __restrict__ b2,
                float* __restrict__ agg,
                long long E, int ntiles) {
  __shared__ _Float16 hid[EWAVES][16 * 64];
  const int lane = threadIdx.x & 31;
  const int wv   = threadIdx.x >> 5;
  const int tile = blockIdx.x * EWAVES + wv;
  const bool active = tile < ntiles;          // wave-uniform: EXEC stays full
  const int  mlo = lane & 15;
  const int  mhi = (lane & 16) ? 8 : 0;
  const long long e0 = (long long)tile * 16;

  if (active) {
    const int drow = (int)eidx[E + e0 + mlo];   // x_i (aggregation target)
    const int srow = (int)eidx[e0 + mlo];       // x_j
    const _Float16* hd = h + (long long)drow * 64;
    const _Float16* hs = h + (long long)srow * 64;
    const float* ea = eattr + (e0 + mlo) * 16;

    v8f acc[4];
    acc_zero(acc);
#pragma unroll
    for (int kt = 0; kt < 5; ++kt) {
      v16h a;
      if (kt < 2) {
        a = afrag_f16g(hd, kt * 32, lane);
      } else if (kt < 4) {
        a = afrag_f16g(hs, (kt - 2) * 32, lane);
      } else {  // cols 128..143 = edge_attr, 144..159 = zero pad
        const int off = (lane & 16) ? 8 : 0;
#pragma unroll
        for (int i = 0; i < 8; ++i) {
          a[i]     = (_Float16)ea[off + i];
          a[i + 8] = (_Float16)0.f;
        }
      }
#pragma unroll
      for (int nt = 0; nt < 4; ++nt)
        acc[nt] = wmma16(a, bfrag(w1f, kt, nt, lane), acc[nt]);
    }
    // bias + relu -> LDS f16 staging (C layout -> row-major tile)
#pragma unroll
    for (int nt = 0; nt < 4; ++nt) {
      const int n = nt * 16 + mlo;
      const float bb = b1[n];
#pragma unroll
      for (int r = 0; r < 8; ++r) {
        float v = acc[nt][r] + bb;
        hid[wv][(mhi + r) * 64 + n] = (_Float16)(v > 0.f ? v : 0.f);
      }
    }
  }
  __syncthreads();
  if (active) {
    v8f acc[4];
    acc_zero(acc);
#pragma unroll
    for (int kt = 0; kt < 2; ++kt) {
      v16h a = afrag_lds(&hid[wv][mlo * 64], kt * 32, lane);
#pragma unroll
      for (int nt = 0; nt < 4; ++nt)
        acc[nt] = wmma16(a, bfrag(w2f, kt, nt, lane), acc[nt]);
    }
    int drows[8];
#pragma unroll
    for (int r = 0; r < 8; ++r) drows[r] = (int)eidx[E + e0 + mhi + r];
#pragma unroll
    for (int nt = 0; nt < 4; ++nt) {
      const int n = nt * 16 + mlo;
      const float bb = b2[n];
#pragma unroll
      for (int r = 0; r < 8; ++r)
        atomicAdd(&agg[(long long)drows[r] * 64 + n], acc[nt][r] + bb);
    }
  }
}

// ---------------------------------------------------------------------------
// Node update kernel: per wave, a tile of 16 nodes. h in/out are f16; agg f32.
//   u_in = [h(64) | agg(64)] ; h' = relu(relu(u_in @ W1 + b1) @ W2 + b2)
// ---------------------------------------------------------------------------
__global__ void __launch_bounds__(NWAVES * 32)
node_upd_kernel(const _Float16* __restrict__ h, const float* __restrict__ agg,
                const _Float16* __restrict__ w1f, const float* __restrict__ b1,
                const _Float16* __restrict__ w2f, const float* __restrict__ b2,
                _Float16* __restrict__ hout, int ntiles) {
  __shared__ _Float16 hid[NWAVES][16 * 64];
  const int lane = threadIdx.x & 31;
  const int wv   = threadIdx.x >> 5;
  const int tile = blockIdx.x * NWAVES + wv;
  const bool active = tile < ntiles;
  const int  mlo = lane & 15;
  const int  mhi = (lane & 16) ? 8 : 0;
  const long long n0 = (long long)tile * 16;

  if (active) {
    const _Float16* hr = h + (n0 + mlo) * 64;
    const float*    ar = agg + (n0 + mlo) * 64;
    v8f acc[4];
    acc_zero(acc);
#pragma unroll
    for (int kt = 0; kt < 4; ++kt) {
      v16h a = (kt < 2) ? afrag_f16g(hr, kt * 32, lane)
                        : afrag_f32(ar, (kt - 2) * 32, lane);
#pragma unroll
      for (int nt = 0; nt < 4; ++nt)
        acc[nt] = wmma16(a, bfrag(w1f, kt, nt, lane), acc[nt]);
    }
#pragma unroll
    for (int nt = 0; nt < 4; ++nt) {
      const int n = nt * 16 + mlo;
      const float bb = b1[n];
#pragma unroll
      for (int r = 0; r < 8; ++r) {
        float v = acc[nt][r] + bb;
        hid[wv][(mhi + r) * 64 + n] = (_Float16)(v > 0.f ? v : 0.f);
      }
    }
  }
  __syncthreads();
  if (active) {
    v8f acc[4];
    acc_zero(acc);
#pragma unroll
    for (int kt = 0; kt < 2; ++kt) {
      v16h a = afrag_lds(&hid[wv][mlo * 64], kt * 32, lane);
#pragma unroll
      for (int nt = 0; nt < 4; ++nt)
        acc[nt] = wmma16(a, bfrag(w2f, kt, nt, lane), acc[nt]);
    }
#pragma unroll
    for (int nt = 0; nt < 4; ++nt) {
      const int n = nt * 16 + mlo;
      const float bb = b2[n];
#pragma unroll
      for (int r = 0; r < 8; ++r) {
        float v = acc[nt][r] + bb;        // inner bias
        v = v > 0.f ? v : 0.f;            // outer relu of model forward
        hout[(n0 + mhi + r) * 64 + n] = (_Float16)v;
      }
    }
  }
}

// ---------------------------------------------------------------------------
// Graph mean-pool (f32 atomic segment sum over f16 h) and tiny head MLP.
// ---------------------------------------------------------------------------
__global__ void pool_kernel(const _Float16* __restrict__ h,
                            const long long* __restrict__ batch,
                            float* __restrict__ pooled, float* __restrict__ counts,
                            long long total) {
  long long t = (long long)blockIdx.x * blockDim.x + threadIdx.x;
  if (t >= total) return;
  const int n = (int)(t >> 6);
  const int c = (int)(t & 63);
  const int g = (int)batch[n];
  atomicAdd(&pooled[(long long)g * 64 + c], (float)h[t]);
  if (c == 0) atomicAdd(&counts[g], 1.f);
}

__global__ void final_kernel(const float* __restrict__ pooled,
                             const float* __restrict__ counts,
                             const float* __restrict__ w1, const float* __restrict__ b1,
                             const float* __restrict__ w2, const float* __restrict__ b2,
                             float* __restrict__ out) {
  __shared__ float red[64];
  const int g = blockIdx.x;
  const int j = threadIdx.x;
  float cnt = counts[g];
  cnt = cnt > 1.f ? cnt : 1.f;
  const float inv = 1.f / cnt;
  float s = 0.f;
#pragma unroll 8
  for (int k = 0; k < 64; ++k)
    s += (pooled[(long long)g * 64 + k] * inv) * w1[k * 64 + j];
  s += b1[j];
  s = s > 0.f ? s : 0.f;
  red[j] = s * w2[j];
  for (int off = 32; off > 0; off >>= 1) {
    __syncthreads();
    if (j < off) red[j] += red[j + off];
  }
  __syncthreads();
  if (j == 0) out[g] = red[0] + b2[0];
}

// ---------------------------------------------------------------------------
// Host-side orchestration
// ---------------------------------------------------------------------------
extern "C" void kernel_launch(void* const* d_in, const int* in_sizes, int n_in,
                              void* d_out, int out_size, void* d_ws, size_t ws_size,
                              hipStream_t stream) {
  (void)n_in; (void)ws_size;
  const float*     x     = (const float*)d_in[0];
  const long long* ei    = (const long long*)d_in[1];
  const float*     eattr = (const float*)d_in[2];
  const long long* batch = (const long long*)d_in[3];
  const float* mW1 = (const float*)d_in[4];
  const float* mb1 = (const float*)d_in[5];
  const float* mW2 = (const float*)d_in[6];
  const float* mb2 = (const float*)d_in[7];
  const float* uW1 = (const float*)d_in[8];
  const float* ub1 = (const float*)d_in[9];
  const float* uW2 = (const float*)d_in[10];
  const float* ub2 = (const float*)d_in[11];
  const float* l1w = (const float*)d_in[12];
  const float* l1b = (const float*)d_in[13];
  const float* l2w = (const float*)d_in[14];
  const float* l2b = (const float*)d_in[15];

  const long long N = in_sizes[0] / 64;
  const long long E = in_sizes[1] / 2;
  const int L = in_sizes[5] / 64;    // mb1 is [L, 64]
  const int G = out_size;            // output is [G, 1]

  // Workspace carve-out (256B aligned slices).
  char* p = (char*)d_ws;
  auto carve = [&](size_t bytes) -> char* {
    char* r = p;
    p += (bytes + 255) & ~(size_t)255;
    return r;
  };
  _Float16* hx     = (_Float16*)carve((size_t)N * 64 * 2);  // f16 copy of x
  _Float16* h0     = (_Float16*)carve((size_t)N * 64 * 2);
  _Float16* h1     = (_Float16*)carve((size_t)N * 64 * 2);
  float*    agg    = (float*)carve((size_t)N * 64 * 4);
  float*    pooled = (float*)carve((size_t)G * 64 * 4);
  float*    counts = (float*)carve((size_t)G * 4);
  _Float16* mW1f = (_Float16*)carve((size_t)L * 5 * 2048 * 2);
  _Float16* mW2f = (_Float16*)carve((size_t)L * 2 * 2048 * 2);
  _Float16* uW1f = (_Float16*)carve((size_t)L * 4 * 2048 * 2);
  _Float16* uW2f = (_Float16*)carve((size_t)L * 2 * 2048 * 2);

  // Pack weights into WMMA B-fragment layout (f16), K zero-padded.
  {
    int tot;
    tot = L * 5 * 2048; pack_weights<<<(tot + 255) / 256, 256, 0, stream>>>(mW1, mW1f, L, 144, 5);
    tot = L * 2 * 2048; pack_weights<<<(tot + 255) / 256, 256, 0, stream>>>(mW2, mW2f, L, 64, 2);
    tot = L * 4 * 2048; pack_weights<<<(tot + 255) / 256, 256, 0, stream>>>(uW1, uW1f, L, 128, 4);
    tot = L * 2 * 2048; pack_weights<<<(tot + 255) / 256, 256, 0, stream>>>(uW2, uW2f, L, 64, 2);
  }

  const int etiles = (int)((E + 15) / 16);
  const int vtiles = (int)((N + 15) / 16);
  const long long hElems = N * 64;

  // x -> f16 once (layer 0 input).
  f32_to_f16<<<(int)((hElems + 255) / 256), 256, 0, stream>>>(x, hx, hElems);

  const _Float16* hin = hx;
  _Float16* bufs[2] = {h0, h1};
  for (int l = 0; l < L; ++l) {
    zero_f32<<<(int)((hElems + 255) / 256), 256, 0, stream>>>(agg, hElems);
    edge_msg_kernel<<<(etiles + EWAVES - 1) / EWAVES, EWAVES * 32, 0, stream>>>(
        hin, ei, eattr,
        mW1f + (size_t)l * 5 * 2048, mb1 + l * 64,
        mW2f + (size_t)l * 2 * 2048, mb2 + l * 64,
        agg, E, etiles);
    _Float16* hout = bufs[l & 1];
    node_upd_kernel<<<(vtiles + NWAVES - 1) / NWAVES, NWAVES * 32, 0, stream>>>(
        hin, agg,
        uW1f + (size_t)l * 4 * 2048, ub1 + l * 64,
        uW2f + (size_t)l * 2 * 2048, ub2 + l * 64,
        hout, vtiles);
    hin = hout;
  }

  zero_f32<<<(int)(((long long)G * 64 + 255) / 256), 256, 0, stream>>>(pooled, (long long)G * 64);
  zero_f32<<<(G + 255) / 256, 256, 0, stream>>>(counts, G);
  pool_kernel<<<(int)((hElems + 255) / 256), 256, 0, stream>>>(hin, batch, pooled, counts, hElems);
  final_kernel<<<G, 64, 0, stream>>>(pooled, counts, l1w, l1b, l2w, l2b, (float*)d_out);
}
